// ProteinMPNN_12094627906365
// MI455X (gfx1250) — matrix operand order repaired and checked by reading, not compiled, not run on hardware
//
#include <hip/hip_runtime.h>

typedef __attribute__((ext_vector_type(16))) __bf16 v16bf;
typedef __attribute__((ext_vector_type(8)))  float  v8f;
typedef __attribute__((ext_vector_type(8)))  unsigned short u16x8;

namespace {
constexpr int Bn = 2, Ln = 2048, Kn = 48, Hh = 128;
constexpr int LDA  = 264;  // 256+8 pad (ushort elems; rows 16B aligned)
constexpr int LDH1 = 136;  // 128+8
constexpr int LDHF = 520;  // 512+8
}

__device__ __forceinline__ unsigned short f32_to_bf16(float f) {
  union { float f; unsigned int u; } v; v.f = f;
  unsigned int r = v.u + 0x7FFFu + ((v.u >> 16) & 1u);
  return (unsigned short)(r >> 16);
}
__device__ __forceinline__ unsigned int pack_bf16x2(float a, float b) {
  return (unsigned int)f32_to_bf16(a) | ((unsigned int)f32_to_bf16(b) << 16);
}
__device__ __forceinline__ float gelu_erf(float x) {
  return 0.5f * x * (1.0f + erff(x * 0.70710678118654752f));
}
// A fragment (16-bit A 16x32 layout): lane m (0-15) K=[kb..kb+7],[kb+16..kb+23];
// lane 16+m: K=[kb+8..kb+15],[kb+24..kb+31]  -> two 16B LDS loads.
__device__ __forceinline__ v16bf lds_a_frag(const unsigned short* rowp, int kb, int hi8) {
  union { u16x8 h[2]; v16bf v; } u;
  u.h[0] = *reinterpret_cast<const u16x8*>(rowp + kb + hi8);
  u.h[1] = *reinterpret_cast<const u16x8*>(rowp + kb + 16 + hi8);
  return u.v;
}
// B fragment: prepacked 32 lanes x 16 bf16 contiguous per (kt,nt) tile.
__device__ __forceinline__ v16bf glb_b_frag(const unsigned short* base, int tile, int lane) {
  return *reinterpret_cast<const v16bf*>(base + (size_t)tile * 512 + (size_t)lane * 16);
}
#define WMMA_BF16(A, Bv, C) \
  __builtin_amdgcn_wmma_f32_16x16x32_bf16(false, (A), false, (Bv), (short)0, (C), false, false)

// ---------------------------------------------------------------------------
// Weight -> bf16 WMMA B-fragment layout. One block per (kt,nt) tile, 32 lanes.
// dense B 32x16: lane n(0-15)=col, h=lane>>4; elems 0-7: K=8h+e, elems 8-15: K=16+8h+(e-8).
__global__ __launch_bounds__(32) void frag_convert(
    const float* __restrict__ W, unsigned short* __restrict__ out,
    int Kdim, int N, int row_off) {
  int tile = blockIdx.x;
  int NT = N >> 4;
  int kt = tile / NT, nt = tile % NT;
  int lane = threadIdx.x;
  int n = nt * 16 + (lane & 15);
  int kbase = kt * 32 + ((lane >> 4) << 3);
  unsigned short* dst = out + (size_t)tile * 512 + (size_t)lane * 16;
#pragma unroll
  for (int e = 0; e < 16; ++e)
    dst[e] = f32_to_bf16(W[(size_t)(row_off + kbase + (e < 8 ? e : e + 8)) * N + n]);
}

// ---------------------------------------------------------------------------
// Node message: one block = one node (48 edges). 4 waves, 128 threads.
__global__ __launch_bounds__(128) void node_msg_kernel(
    const float* __restrict__ hV, const float* __restrict__ hE,
    const int* __restrict__ Eidx, const float* __restrict__ maskA,
    const float* __restrict__ W1, const float* __restrict__ b1,
    const unsigned short* __restrict__ W1f, const unsigned short* __restrict__ W2f,
    const unsigned short* __restrict__ W3f,
    const float* __restrict__ b2, const float* __restrict__ b3,
    const float* __restrict__ g1, const float* __restrict__ be1,
    float* __restrict__ hVmid) {
  __shared__ alignas(16) unsigned short Abuf[Kn * LDA];   // 25,344 B (reused for h2)
  __shared__ alignas(16) unsigned short h1buf[Kn * LDH1]; // 13,056 B
  __shared__ float hv_s[Hh];
  __shared__ float hvW[Hh];
  __shared__ int   idx_s[Kn];
  __shared__ float mask_s[Kn];
  __shared__ float dh_part[2 * Hh];
  __shared__ float red[Hh], red2[Hh];

  const int bl = blockIdx.x;
  const int b  = bl / Ln;
  const int t = threadIdx.x, wave = t >> 5, lane = t & 31;
  const int m15 = lane & 15, hi8 = (lane >> 4) << 3;

  if (t < Kn) { idx_s[t] = Eidx[(size_t)bl * Kn + t]; mask_s[t] = maskA[(size_t)bl * Kn + t]; }
  hv_s[t] = hV[(size_t)bl * Hh + t];
  __syncthreads();

  // self-part of layer 1: hvW = b1 + hv @ W1[0:128,:]
  {
    float a = b1[t];
#pragma unroll 4
    for (int j = 0; j < Hh; ++j) a += hv_s[j] * W1[(size_t)j * Hh + t];
    hvW[t] = a;
  }
  // stage A = [bf16(h_E) | bf16(h_nb)]  (48 x 256)
#pragma unroll 2
  for (int i = 0; i < 12; ++i) {
    int row = wave * 12 + i;
    float4 e4 = *reinterpret_cast<const float4*>(&hE[(((size_t)bl * Kn) + row) * Hh + lane * 4]);
    uint2 p0; p0.x = pack_bf16x2(e4.x, e4.y); p0.y = pack_bf16x2(e4.z, e4.w);
    *reinterpret_cast<uint2*>(&Abuf[row * LDA + lane * 4]) = p0;
    int nb = idx_s[row];
    float4 n4 = *reinterpret_cast<const float4*>(&hV[((size_t)b * Ln + nb) * Hh + lane * 4]);
    uint2 p1; p1.x = pack_bf16x2(n4.x, n4.y); p1.y = pack_bf16x2(n4.z, n4.w);
    *reinterpret_cast<uint2*>(&Abuf[row * LDA + 128 + lane * 4]) = p1;
  }
  __syncthreads();

  const v8f zero8 = {0.f,0.f,0.f,0.f,0.f,0.f,0.f,0.f};

  // ---- layer 1: (48x256)@(256x128) ----
  v8f acc1[3][2];
#pragma unroll
  for (int mt = 0; mt < 3; ++mt)
#pragma unroll
    for (int j = 0; j < 2; ++j) acc1[mt][j] = zero8;
  for (int kt = 0; kt < 8; ++kt) {
    int kb = kt * 32;
    if (kt < 7)
      __builtin_prefetch(W1f + ((size_t)((kt + 1) * 8 + wave * 2) * 512), 0, 3);
    v16bf a[3];
#pragma unroll
    for (int mt = 0; mt < 3; ++mt)
      a[mt] = lds_a_frag(&Abuf[(mt * 16 + m15) * LDA], kb, hi8);
#pragma unroll
    for (int j = 0; j < 2; ++j) {
      v16bf bf = glb_b_frag(W1f, kt * 8 + wave * 2 + j, lane);
#pragma unroll
      for (int mt = 0; mt < 3; ++mt) acc1[mt][j] = WMMA_BF16(a[mt], bf, acc1[mt][j]);
    }
  }
#pragma unroll
  for (int j = 0; j < 2; ++j) {
    int n = (wave * 2 + j) * 16 + m15;
    float addv = hvW[n];
#pragma unroll
    for (int mt = 0; mt < 3; ++mt) {
      int rbase = mt * 16 + hi8;
#pragma unroll
      for (int r = 0; r < 8; ++r)
        h1buf[(rbase + r) * LDH1 + n] = f32_to_bf16(gelu_erf(acc1[mt][j][r] + addv));
    }
  }
  __syncthreads();

  // ---- layer 2: (48x128)@(128x128), h2 -> Abuf (reuse) ----
  unsigned short* h2buf = Abuf;
  v8f acc2[3][2];
#pragma unroll
  for (int mt = 0; mt < 3; ++mt)
#pragma unroll
    for (int j = 0; j < 2; ++j) acc2[mt][j] = zero8;
  for (int kt = 0; kt < 4; ++kt) {
    int kb = kt * 32;
    v16bf a[3];
#pragma unroll
    for (int mt = 0; mt < 3; ++mt)
      a[mt] = lds_a_frag(&h1buf[(mt * 16 + m15) * LDH1], kb, hi8);
#pragma unroll
    for (int j = 0; j < 2; ++j) {
      v16bf bf = glb_b_frag(W2f, kt * 8 + wave * 2 + j, lane);
#pragma unroll
      for (int mt = 0; mt < 3; ++mt) acc2[mt][j] = WMMA_BF16(a[mt], bf, acc2[mt][j]);
    }
  }
#pragma unroll
  for (int j = 0; j < 2; ++j) {
    int n = (wave * 2 + j) * 16 + m15;
    float bb = b2[n];
#pragma unroll
    for (int mt = 0; mt < 3; ++mt) {
      int rbase = mt * 16 + hi8;
#pragma unroll
      for (int r = 0; r < 8; ++r)
        h2buf[(rbase + r) * LDH1 + n] = f32_to_bf16(gelu_erf(acc2[mt][j][r] + bb));
    }
  }
  __syncthreads();

  // ---- layer 3 + mask + K-reduction ----
  v8f acc3[3][2];
#pragma unroll
  for (int mt = 0; mt < 3; ++mt)
#pragma unroll
    for (int j = 0; j < 2; ++j) acc3[mt][j] = zero8;
  for (int kt = 0; kt < 4; ++kt) {
    int kb = kt * 32;
    v16bf a[3];
#pragma unroll
    for (int mt = 0; mt < 3; ++mt)
      a[mt] = lds_a_frag(&h2buf[(mt * 16 + m15) * LDH1], kb, hi8);
#pragma unroll
    for (int j = 0; j < 2; ++j) {
      v16bf bf = glb_b_frag(W3f, kt * 8 + wave * 2 + j, lane);
#pragma unroll
      for (int mt = 0; mt < 3; ++mt) acc3[mt][j] = WMMA_BF16(a[mt], bf, acc3[mt][j]);
    }
  }
#pragma unroll
  for (int j = 0; j < 2; ++j) {
    int n = (wave * 2 + j) * 16 + m15;
    float bb = b3[n];
    float colsum = 0.f;
#pragma unroll
    for (int mt = 0; mt < 3; ++mt) {
      int rbase = mt * 16 + hi8;
#pragma unroll
      for (int r = 0; r < 8; ++r)
        colsum += mask_s[rbase + r] * (acc3[mt][j][r] + bb);
    }
    dh_part[n * 2 + (lane >> 4)] = colsum;
  }
  __syncthreads();

  // residual + LayerNorm (block of 128 threads == 128 features)
  float x = hv_s[t] + (dh_part[t * 2] + dh_part[t * 2 + 1]) * (1.0f / 30.0f);
  red[t] = x; red2[t] = x * x;
  __syncthreads();
  for (int s = 64; s > 0; s >>= 1) {
    if (t < s) { red[t] += red[t + s]; red2[t] += red2[t + s]; }
    __syncthreads();
  }
  float mu  = red[0] * (1.0f / 128.0f);
  float var = red2[0] * (1.0f / 128.0f) - mu * mu;
  hVmid[(size_t)bl * Hh + t] = g1[t] * (x - mu) * rsqrtf(var + 1e-5f) + be1[t];
}

// ---------------------------------------------------------------------------
// FFN + LN2 + mask_V: one block = 16 nodes.
__global__ __launch_bounds__(128) void ffn_kernel(
    const float* __restrict__ hVmid, const float* __restrict__ maskV,
    const unsigned short* __restrict__ Winf, const float* __restrict__ b_in,
    const unsigned short* __restrict__ Woutf, const float* __restrict__ b_out,
    const float* __restrict__ g2, const float* __restrict__ be2,
    float* __restrict__ out_hV, unsigned short* __restrict__ hVnewbf) {
  __shared__ alignas(16) unsigned short Abuf2[16 * LDH1];
  __shared__ alignas(16) unsigned short Hbuf[16 * LDHF];
  __shared__ float ybuf[16 * Hh];

  const int node0 = blockIdx.x * 16;
  const int t = threadIdx.x, wave = t >> 5, lane = t & 31;
  const int m15 = lane & 15, hi8 = (lane >> 4) << 3;

#pragma unroll
  for (int i = 0; i < 16; ++i)
    Abuf2[i * LDH1 + t] = f32_to_bf16(hVmid[((size_t)node0 + i) * Hh + t]);
  __syncthreads();

  const v8f zero8 = {0.f,0.f,0.f,0.f,0.f,0.f,0.f,0.f};

  // ---- in: (16x128)@(128x512), each wave 8 N-tiles ----
  v8f accA[8];
#pragma unroll
  for (int j = 0; j < 8; ++j) accA[j] = zero8;
  for (int kt = 0; kt < 4; ++kt) {
    int kb = kt * 32;
    if (kt < 3)
      __builtin_prefetch(Winf + ((size_t)((kt + 1) * 32 + wave * 8) * 512), 0, 3);
    v16bf a = lds_a_frag(&Abuf2[m15 * LDH1], kb, hi8);
#pragma unroll
    for (int j = 0; j < 8; ++j) {
      v16bf bf = glb_b_frag(Winf, kt * 32 + wave * 8 + j, lane);
      accA[j] = WMMA_BF16(a, bf, accA[j]);
    }
  }
#pragma unroll
  for (int j = 0; j < 8; ++j) {
    int n = (wave * 8 + j) * 16 + m15;
    float bb = b_in[n];
#pragma unroll
    for (int r = 0; r < 8; ++r)
      Hbuf[(hi8 + r) * LDHF + n] = f32_to_bf16(gelu_erf(accA[j][r] + bb));
  }
  __syncthreads();

  // ---- out: (16x512)@(512x128), each wave 2 N-tiles ----
  v8f accO[2];
  accO[0] = zero8; accO[1] = zero8;
  for (int kt = 0; kt < 16; ++kt) {
    int kb = kt * 32;
    v16bf a = lds_a_frag(&Hbuf[m15 * LDHF], kb, hi8);
#pragma unroll
    for (int j = 0; j < 2; ++j) {
      v16bf bf = glb_b_frag(Woutf, kt * 8 + wave * 2 + j, lane);
      accO[j] = WMMA_BF16(a, bf, accO[j]);
    }
  }
#pragma unroll
  for (int j = 0; j < 2; ++j) {
    int n = (wave * 2 + j) * 16 + m15;
    float bb = b_out[n];
#pragma unroll
    for (int r = 0; r < 8; ++r)
      ybuf[(hi8 + r) * Hh + n] = accO[j][r] + bb;
  }
  __syncthreads();

  // residual + per-row LN + mask_V; wave handles rows wave, wave+4, ...
#pragma unroll
  for (int i = 0; i < 4; ++i) {
    int row = wave + 4 * i;
    size_t nd = (size_t)node0 + row;
    float4 hv4 = *reinterpret_cast<const float4*>(&hVmid[nd * Hh + lane * 4]);
    float xs[4];
    xs[0] = hv4.x + ybuf[row * Hh + lane * 4 + 0];
    xs[1] = hv4.y + ybuf[row * Hh + lane * 4 + 1];
    xs[2] = hv4.z + ybuf[row * Hh + lane * 4 + 2];
    xs[3] = hv4.w + ybuf[row * Hh + lane * 4 + 3];
    float s1 = xs[0] + xs[1] + xs[2] + xs[3];
    float s2 = xs[0]*xs[0] + xs[1]*xs[1] + xs[2]*xs[2] + xs[3]*xs[3];
#pragma unroll
    for (int off = 16; off > 0; off >>= 1) {
      s1 += __shfl_xor(s1, off, 32);
      s2 += __shfl_xor(s2, off, 32);
    }
    float mu = s1 * (1.0f / 128.0f);
    float var = s2 * (1.0f / 128.0f) - mu * mu;
    float rs = rsqrtf(var + 1e-5f);
    float mv = maskV[nd];
#pragma unroll
    for (int c = 0; c < 4; ++c) {
      int col = lane * 4 + c;
      float o = mv * (g2[col] * (xs[c] - mu) * rs + be2[col]);
      out_hV[nd * Hh + col] = o;
      hVnewbf[nd * Hh + col] = f32_to_bf16(o);
    }
  }
}

// ---------------------------------------------------------------------------
// Edge update: one block = one node's 48 edges.
__global__ __launch_bounds__(128) void edge_kernel(
    const float* __restrict__ hVout, const unsigned short* __restrict__ hVnewbf,
    const float* __restrict__ hE, const int* __restrict__ Eidx,
    const float* __restrict__ W11, const float* __restrict__ b11,
    const unsigned short* __restrict__ W11f, const unsigned short* __restrict__ W12f,
    const unsigned short* __restrict__ W13f,
    const float* __restrict__ b12, const float* __restrict__ b13,
    const float* __restrict__ g3, const float* __restrict__ be3,
    float* __restrict__ out_hE) {
  __shared__ alignas(16) unsigned short Abuf[Kn * LDA];   // reused: h2, then xbuf(f32)
  __shared__ alignas(16) unsigned short h1buf[Kn * LDH1];
  __shared__ float hv_s[Hh];
  __shared__ float hvW[Hh];
  __shared__ int   idx_s[Kn];

  const int bl = blockIdx.x;
  const int b  = bl / Ln;
  const int t = threadIdx.x, wave = t >> 5, lane = t & 31;
  const int m15 = lane & 15, hi8 = (lane >> 4) << 3;

  if (t < Kn) idx_s[t] = Eidx[(size_t)bl * Kn + t];
  hv_s[t] = hVout[(size_t)bl * Hh + t];
  __syncthreads();

  {
    float a = b11[t];
#pragma unroll 4
    for (int j = 0; j < Hh; ++j) a += hv_s[j] * W11[(size_t)j * Hh + t];
    hvW[t] = a;
  }
#pragma unroll 2
  for (int i = 0; i < 12; ++i) {
    int row = wave * 12 + i;
    float4 e4 = *reinterpret_cast<const float4*>(&hE[(((size_t)bl * Kn) + row) * Hh + lane * 4]);
    uint2 p0; p0.x = pack_bf16x2(e4.x, e4.y); p0.y = pack_bf16x2(e4.z, e4.w);
    *reinterpret_cast<uint2*>(&Abuf[row * LDA + lane * 4]) = p0;
    int nb = idx_s[row];
    uint2 p1 = *reinterpret_cast<const uint2*>(&hVnewbf[((size_t)b * Ln + nb) * Hh + lane * 4]);
    *reinterpret_cast<uint2*>(&Abuf[row * LDA + 128 + lane * 4]) = p1;
  }
  __syncthreads();

  const v8f zero8 = {0.f,0.f,0.f,0.f,0.f,0.f,0.f,0.f};

  v8f acc1[3][2];
#pragma unroll
  for (int mt = 0; mt < 3; ++mt)
#pragma unroll
    for (int j = 0; j < 2; ++j) acc1[mt][j] = zero8;
  for (int kt = 0; kt < 8; ++kt) {
    int kb = kt * 32;
    if (kt < 7)
      __builtin_prefetch(W11f + ((size_t)((kt + 1) * 8 + wave * 2) * 512), 0, 3);
    v16bf a[3];
#pragma unroll
    for (int mt = 0; mt < 3; ++mt)
      a[mt] = lds_a_frag(&Abuf[(mt * 16 + m15) * LDA], kb, hi8);
#pragma unroll
    for (int j = 0; j < 2; ++j) {
      v16bf bf = glb_b_frag(W11f, kt * 8 + wave * 2 + j, lane);
#pragma unroll
      for (int mt = 0; mt < 3; ++mt) acc1[mt][j] = WMMA_BF16(a[mt], bf, acc1[mt][j]);
    }
  }
#pragma unroll
  for (int j = 0; j < 2; ++j) {
    int n = (wave * 2 + j) * 16 + m15;
    float addv = hvW[n];
#pragma unroll
    for (int mt = 0; mt < 3; ++mt) {
      int rbase = mt * 16 + hi8;
#pragma unroll
      for (int r = 0; r < 8; ++r)
        h1buf[(rbase + r) * LDH1 + n] = f32_to_bf16(gelu_erf(acc1[mt][j][r] + addv));
    }
  }
  __syncthreads();

  unsigned short* h2buf = Abuf;
  v8f acc2[3][2];
#pragma unroll
  for (int mt = 0; mt < 3; ++mt)
#pragma unroll
    for (int j = 0; j < 2; ++j) acc2[mt][j] = zero8;
  for (int kt = 0; kt < 4; ++kt) {
    int kb = kt * 32;
    v16bf a[3];
#pragma unroll
    for (int mt = 0; mt < 3; ++mt)
      a[mt] = lds_a_frag(&h1buf[(mt * 16 + m15) * LDH1], kb, hi8);
#pragma unroll
    for (int j = 0; j < 2; ++j) {
      v16bf bf = glb_b_frag(W12f, kt * 8 + wave * 2 + j, lane);
#pragma unroll
      for (int mt = 0; mt < 3; ++mt) acc2[mt][j] = WMMA_BF16(a[mt], bf, acc2[mt][j]);
    }
  }
#pragma unroll
  for (int j = 0; j < 2; ++j) {
    int n = (wave * 2 + j) * 16 + m15;
    float bb = b12[n];
#pragma unroll
    for (int mt = 0; mt < 3; ++mt) {
      int rbase = mt * 16 + hi8;
#pragma unroll
      for (int r = 0; r < 8; ++r)
        h2buf[(rbase + r) * LDH1 + n] = f32_to_bf16(gelu_erf(acc2[mt][j][r] + bb));
    }
  }
  __syncthreads();

  v8f acc3[3][2];
#pragma unroll
  for (int mt = 0; mt < 3; ++mt)
#pragma unroll
    for (int j = 0; j < 2; ++j) acc3[mt][j] = zero8;
  for (int kt = 0; kt < 4; ++kt) {
    int kb = kt * 32;
    v16bf a[3];
#pragma unroll
    for (int mt = 0; mt < 3; ++mt)
      a[mt] = lds_a_frag(&h2buf[(mt * 16 + m15) * LDH1], kb, hi8);
#pragma unroll
    for (int j = 0; j < 2; ++j) {
      v16bf bf = glb_b_frag(W13f, kt * 8 + wave * 2 + j, lane);
#pragma unroll
      for (int mt = 0; mt < 3; ++mt) acc3[mt][j] = WMMA_BF16(a[mt], bf, acc3[mt][j]);
    }
  }
  __syncthreads();  // everyone done reading h2buf (aliases Abuf)

  float* xbuf = reinterpret_cast<float*>(Abuf);  // 48 x 128 msg values
#pragma unroll
  for (int j = 0; j < 2; ++j) {
    int n = (wave * 2 + j) * 16 + m15;
    float bb = b13[n];
#pragma unroll
    for (int mt = 0; mt < 3; ++mt) {
      int rbase = mt * 16 + hi8;
#pragma unroll
      for (int r = 0; r < 8; ++r)
        xbuf[(rbase + r) * Hh + n] = acc3[mt][j][r] + bb;
    }
  }
  __syncthreads();

  // residual (h_E) + per-row LN; wave handles rows wave, wave+4, ... (12 rows)
#pragma unroll 2
  for (int i = 0; i < 12; ++i) {
    int row = wave + 4 * i;
    float4 e4 = *reinterpret_cast<const float4*>(&hE[(((size_t)bl * Kn) + row) * Hh + lane * 4]);
    float xs[4];
    xs[0] = e4.x + xbuf[row * Hh + lane * 4 + 0];
    xs[1] = e4.y + xbuf[row * Hh + lane * 4 + 1];
    xs[2] = e4.z + xbuf[row * Hh + lane * 4 + 2];
    xs[3] = e4.w + xbuf[row * Hh + lane * 4 + 3];
    float s1 = xs[0] + xs[1] + xs[2] + xs[3];
    float s2 = xs[0]*xs[0] + xs[1]*xs[1] + xs[2]*xs[2] + xs[3]*xs[3];
#pragma unroll
    for (int off = 16; off > 0; off >>= 1) {
      s1 += __shfl_xor(s1, off, 32);
      s2 += __shfl_xor(s2, off, 32);
    }
    float mu = s1 * (1.0f / 128.0f);
    float var = s2 * (1.0f / 128.0f) - mu * mu;
    float rs = rsqrtf(var + 1e-5f);
#pragma unroll
    for (int c = 0; c < 4; ++c) {
      int col = lane * 4 + c;
      out_hE[((size_t)bl * Kn + row) * Hh + col] = g3[col] * (xs[c] - mu) * rs + be3[col];
    }
  }
}

// ---------------------------------------------------------------------------
extern "C" void kernel_launch(void* const* d_in, const int* in_sizes, int n_in,
                              void* d_out, int out_size, void* d_ws, size_t ws_size,
                              hipStream_t stream) {
  const float* hV    = (const float*)d_in[0];
  const float* hE    = (const float*)d_in[1];
  const int*   Eidx  = (const int*)d_in[2];
  const float* maskV = (const float*)d_in[3];
  const float* maskA = (const float*)d_in[4];
  const float* W1  = (const float*)d_in[5];  const float* b1  = (const float*)d_in[6];
  const float* W2  = (const float*)d_in[7];  const float* b2  = (const float*)d_in[8];
  const float* W3  = (const float*)d_in[9];  const float* b3  = (const float*)d_in[10];
  const float* W11 = (const float*)d_in[11]; const float* b11 = (const float*)d_in[12];
  const float* W12 = (const float*)d_in[13]; const float* b12 = (const float*)d_in[14];
  const float* W13 = (const float*)d_in[15]; const float* b13 = (const float*)d_in[16];
  const float* g1  = (const float*)d_in[17]; const float* be1 = (const float*)d_in[18];
  const float* g2  = (const float*)d_in[19]; const float* be2 = (const float*)d_in[20];
  const float* g3  = (const float*)d_in[21]; const float* be3 = (const float*)d_in[22];
  const float* Win = (const float*)d_in[23]; const float* b_in = (const float*)d_in[24];
  const float* Wou = (const float*)d_in[25]; const float* b_out = (const float*)d_in[26];

  char* ws = (char*)d_ws;
  unsigned short* W1f  = (unsigned short*)(ws + 0);        // 256x128 bf16 = 65536 B
  unsigned short* W2f  = (unsigned short*)(ws + 65536);    // 32768 B
  unsigned short* W3f  = (unsigned short*)(ws + 98304);    // 32768 B
  unsigned short* W11f = (unsigned short*)(ws + 131072);   // 65536 B
  unsigned short* W12f = (unsigned short*)(ws + 196608);   // 32768 B
  unsigned short* W13f = (unsigned short*)(ws + 229376);   // 32768 B
  unsigned short* Winf = (unsigned short*)(ws + 262144);   // 131072 B
  unsigned short* Woutf= (unsigned short*)(ws + 393216);   // 131072 B
  float*          hVmid= (float*)(ws + 524288);            // 4096*128 f32
  unsigned short* hVnb = (unsigned short*)(ws + 2621440);  // 4096*128 bf16

  float* out_hV = (float*)d_out;
  float* out_hE = out_hV + (size_t)Bn * Ln * Hh;

  // weight prepack (tiny; L2-resident)
  frag_convert<<<64, 32, 0, stream>>>(W1,  W1f, 256, 128, 128);
  frag_convert<<<32, 32, 0, stream>>>(W2,  W2f, 128, 128, 0);
  frag_convert<<<32, 32, 0, stream>>>(W3,  W3f, 128, 128, 0);
  frag_convert<<<64, 32, 0, stream>>>(W11, W11f, 256, 128, 128);
  frag_convert<<<32, 32, 0, stream>>>(W12, W12f, 128, 128, 0);
  frag_convert<<<32, 32, 0, stream>>>(W13, W13f, 128, 128, 0);
  frag_convert<<<128, 32, 0, stream>>>(Win, Winf, 128, 512, 0);
  frag_convert<<<128, 32, 0, stream>>>(Wou, Woutf, 512, 128, 0);

  node_msg_kernel<<<Bn * Ln, 128, 0, stream>>>(
      hV, hE, Eidx, maskA, W1, b1, W1f, W2f, W3f, b2, b3, g1, be1, hVmid);

  ffn_kernel<<<(Bn * Ln) / 16, 128, 0, stream>>>(
      hVmid, maskV, Winf, b_in, Woutf, b_out, g2, be2, out_hV, hVnb);

  edge_kernel<<<Bn * Ln, 128, 0, stream>>>(
      out_hV, hVnb, hE, Eidx, W11, b11, W11f, W12f, W13f, b12, b13, g3, be3, out_hE);
}
